// WMDMetric_18580028522636
// MI455X (gfx1250) — matrix operand order repaired and checked by reading, not compiled, not run on hardware
//
#include <hip/hip_runtime.h>
#include <math.h>

typedef __attribute__((ext_vector_type(8)))  float  v8f;
typedef __attribute__((ext_vector_type(8)))  __bf16 v8bf;
typedef __attribute__((ext_vector_type(16))) __bf16 v16bf;

union FragBF {
    v16bf v;
    v8bf  h[2];
};

// Problem dims (fixed by the reference): L*B = 64, t1 = t2 = 1024, h = 128.
constexpr int NB = 64;
constexpr int T1 = 1024;
constexpr int T2 = 1024;
constexpr int H  = 128;
constexpr int TILE = 128;            // 128x128 output tile per block
constexpr int SB   = H + 8;          // bf16 LDS row stride (136 elems = 272 B): 4-bank rotate/row

__device__ __forceinline__ unsigned short f2bf_rne(float f) {
    unsigned int u = __float_as_uint(f);
    unsigned int r = u + 0x7FFFu + ((u >> 16) & 1u);   // round-to-nearest-even
    return (unsigned short)(r >> 16);
}
__device__ __forceinline__ float bf2f(unsigned short h) {
    return __uint_as_float(((unsigned int)h) << 16);
}
// Split two floats into packed bf16 hi/lo words (x = hi + lo, |lo| <= 0.5 ulp(hi)).
__device__ __forceinline__ void split2(float f0, float f1,
                                       unsigned int& hi, unsigned int& lo) {
    unsigned short h0 = f2bf_rne(f0), h1 = f2bf_rne(f1);
    unsigned short l0 = f2bf_rne(f0 - bf2f(h0));
    unsigned short l1 = f2bf_rne(f1 - bf2f(h1));
    hi = (unsigned int)h0 | ((unsigned int)h1 << 16);
    lo = (unsigned int)l0 | ((unsigned int)l1 << 16);
}

// ---------------------------------------------------------------------------
// Kernel 0: seed the row/col min workspace with +inf bit patterns each launch.
// ---------------------------------------------------------------------------
__global__ void wmd_init_mins(unsigned int* __restrict__ mins, int n) {
    int i = blockIdx.x * blockDim.x + threadIdx.x;
    if (i < n) mins[i] = 0x7F800000u;   // +inf (d2 >= 0 so uint order == float order)
}

// ---------------------------------------------------------------------------
// Kernel 1: 128x128 block of X·Yᵀ per (batch,tile) via bf16x3-split
// V_WMMA_F32_16X16X32_BF16 (f32-accurate), fused with d2 clamp + row/col mins.
// ---------------------------------------------------------------------------
__global__ __launch_bounds__(256)
void wmd_tile_kernel(const float* __restrict__ x, const float* __restrict__ y,
                     unsigned int* __restrict__ rowmin,   // [NB*T1] min_j d2
                     unsigned int* __restrict__ colmin) { // [NB*T2] min_i d2
    __shared__ __align__(16) unsigned short xh[TILE * SB];   // 34 KB each
    __shared__ __align__(16) unsigned short xl[TILE * SB];
    __shared__ __align__(16) unsigned short yh[TILE * SB];
    __shared__ __align__(16) unsigned short yl[TILE * SB];   // 136 KB bf16 tiles total
    __shared__ float x2s[TILE];
    __shared__ float y2s[TILE];
    __shared__ unsigned int rmin_s[TILE];
    __shared__ unsigned int cmin_s[TILE];

    const int tid     = threadIdx.x;
    const int batch   = blockIdx.x >> 6;
    const int tileIdx = blockIdx.x & 63;
    const int tileM   = tileIdx >> 3;
    const int tileN   = tileIdx & 7;

    const float* xb = x + ((size_t)batch * T1 + (size_t)tileM * TILE) * H;
    const float* yb = y + ((size_t)batch * T2 + (size_t)tileN * TILE) * H;

    if (tid < TILE) rmin_s[tid] = 0x7F800000u;
    else            cmin_s[tid - TILE] = 0x7F800000u;

    // Stage tiles: f32 global -> bf16 hi/lo LDS (RNE split), b128 global loads.
    {
        const float4* xg = (const float4*)xb;
        const float4* yg = (const float4*)yb;
        for (int idx = tid; idx < TILE * H / 4; idx += 256) {
            const int r = idx >> 5;                 // 32 float4 per row
            const int c = (idx & 31) << 2;
            float4 vx = xg[idx];
            float4 vy = yg[idx];
            unsigned int h01, l01, h23, l23;
            split2(vx.x, vx.y, h01, l01);
            split2(vx.z, vx.w, h23, l23);
            *(uint2*)&xh[r * SB + c] = make_uint2(h01, h23);
            *(uint2*)&xl[r * SB + c] = make_uint2(l01, l23);
            split2(vy.x, vy.y, h01, l01);
            split2(vy.z, vy.w, h23, l23);
            *(uint2*)&yh[r * SB + c] = make_uint2(h01, h23);
            *(uint2*)&yl[r * SB + c] = make_uint2(l01, l23);
        }
    }
    __syncthreads();

    // Per-row squared norms from the reconstructed (hi+lo) values.
    if (tid < TILE) {
        const unsigned short* ph = &xh[tid * SB];
        const unsigned short* pl = &xl[tid * SB];
        float s = 0.f;
        #pragma unroll 8
        for (int k = 0; k < H; ++k) { float f = bf2f(ph[k]) + bf2f(pl[k]); s += f * f; }
        x2s[tid] = s;
    } else {
        const unsigned short* ph = &yh[(tid - TILE) * SB];
        const unsigned short* pl = &yl[(tid - TILE) * SB];
        float s = 0.f;
        #pragma unroll 8
        for (int k = 0; k < H; ++k) { float f = bf2f(ph[k]) + bf2f(pl[k]); s += f * f; }
        y2s[tid - TILE] = s;
    }
    __syncthreads();

    // Wave tiling: 8 waves; 4 across M (32 rows each), 2 across N (64 cols each).
    const int lane  = tid & 31;
    const int wave  = tid >> 5;
    const int waveM = wave & 3;
    const int waveN = wave >> 2;
    const int half  = lane >> 4;
    const int ln    = lane & 15;

    v8f acc[2][4];
    #pragma unroll
    for (int mt = 0; mt < 2; ++mt)
        #pragma unroll
        for (int nt = 0; nt < 4; ++nt)
            acc[mt][nt] = (v8f){0.f,0.f,0.f,0.f,0.f,0.f,0.f,0.f};

    // K loop: 16x16x32 slabs; bf16 A frag (16x32): halves hold K {0..7,16..23} /
    // {8..15,24..31}; B frag (32x16): halves hold K {0..7,16..23} / {8..15,24..31}.
    #pragma unroll
    for (int k0 = 0; k0 < H; k0 += 32) {
        FragBF ah[2], al[2], bh[4], bl[4];
        #pragma unroll
        for (int mt = 0; mt < 2; ++mt) {
            const int off = (waveM * 32 + mt * 16 + ln) * SB + k0 + half * 8;
            ah[mt].h[0] = *(const v8bf*)&xh[off];
            ah[mt].h[1] = *(const v8bf*)&xh[off + 16];
            al[mt].h[0] = *(const v8bf*)&xl[off];
            al[mt].h[1] = *(const v8bf*)&xl[off + 16];
        }
        #pragma unroll
        for (int nt = 0; nt < 4; ++nt) {
            const int off = (waveN * 64 + nt * 16 + ln) * SB + k0 + half * 8;
            bh[nt].h[0] = *(const v8bf*)&yh[off];
            bh[nt].h[1] = *(const v8bf*)&yh[off + 16];
            bl[nt].h[0] = *(const v8bf*)&yl[off];
            bl[nt].h[1] = *(const v8bf*)&yl[off + 16];
        }
        #pragma unroll
        for (int mt = 0; mt < 2; ++mt)
            #pragma unroll
            for (int nt = 0; nt < 4; ++nt) {
                // bf16x3 split product: hi*hi + hi*lo + lo*hi (f32 accumulate)
                acc[mt][nt] = __builtin_amdgcn_wmma_f32_16x16x32_bf16(
                    false, ah[mt].v, false, bh[nt].v, (short)0, acc[mt][nt], false, false);
                acc[mt][nt] = __builtin_amdgcn_wmma_f32_16x16x32_bf16(
                    false, ah[mt].v, false, bl[nt].v, (short)0, acc[mt][nt], false, false);
                acc[mt][nt] = __builtin_amdgcn_wmma_f32_16x16x32_bf16(
                    false, al[mt].v, false, bh[nt].v, (short)0, acc[mt][nt], false, false);
            }
    }

    // Epilogue: d2 = x2 + y2 - 2*xy, clamp >= 0; min over rows/cols.
    const float finf = __uint_as_float(0x7F800000u);
    #pragma unroll
    for (int mt = 0; mt < 2; ++mt) {
        float rm[8];
        #pragma unroll
        for (int v = 0; v < 8; ++v) rm[v] = finf;
        #pragma unroll
        for (int nt = 0; nt < 4; ++nt) {
            const int colLocal = waveN * 64 + nt * 16 + ln;
            const float yq = y2s[colLocal];
            float cm = finf;
            #pragma unroll
            for (int v = 0; v < 8; ++v) {
                const int rowLocal = waveM * 32 + mt * 16 + v + half * 8;
                float d2 = fmaxf(x2s[rowLocal] + yq - 2.0f * acc[mt][nt][v], 0.0f);
                rm[v] = fminf(rm[v], d2);
                cm    = fminf(cm, d2);
            }
            atomicMin(&cmin_s[colLocal], __float_as_uint(cm));  // both halves merge
        }
        #pragma unroll
        for (int v = 0; v < 8; ++v) {
            float r = rm[v];
            r = fminf(r, __shfl_xor(r, 1, 32));
            r = fminf(r, __shfl_xor(r, 2, 32));
            r = fminf(r, __shfl_xor(r, 4, 32));
            r = fminf(r, __shfl_xor(r, 8, 32));
            if (ln == 0) {
                const int rowLocal = waveM * 32 + mt * 16 + v + half * 8;
                atomicMin(&rmin_s[rowLocal], __float_as_uint(r));
            }
        }
    }
    __syncthreads();

    // One global atomic per row/col per block.
    if (tid < TILE) {
        atomicMin(&rowmin[(size_t)batch * T1 + tileM * TILE + tid], rmin_s[tid]);
    } else {
        const int t = tid - TILE;
        atomicMin(&colmin[(size_t)batch * T2 + tileN * TILE + t], cmin_s[t]);
    }
}

// ---------------------------------------------------------------------------
// Kernel 2: sum(sqrt(min_d2)) over both axes; out = max(fwd, bwd) / size.
// ---------------------------------------------------------------------------
__global__ __launch_bounds__(1024)
void wmd_reduce_kernel(const unsigned int* __restrict__ rowmin,
                       const unsigned int* __restrict__ colmin,
                       float* __restrict__ out) {
    __shared__ float sr[32];
    __shared__ float sc[32];
    const int tid = threadIdx.x;
    float sumR = 0.f, sumC = 0.f;
    for (int i = tid; i < NB * T1; i += 1024) {
        sumR += sqrtf(__uint_as_float(rowmin[i]));
        sumC += sqrtf(__uint_as_float(colmin[i]));
    }
    #pragma unroll
    for (int o = 16; o > 0; o >>= 1) {
        sumR += __shfl_down(sumR, o, 32);
        sumC += __shfl_down(sumC, o, 32);
    }
    if ((tid & 31) == 0) { sr[tid >> 5] = sumR; sc[tid >> 5] = sumC; }
    __syncthreads();
    if (tid < 32) {
        sumR = sr[tid];
        sumC = sc[tid];
        #pragma unroll
        for (int o = 16; o > 0; o >>= 1) {
            sumR += __shfl_down(sumR, o, 32);
            sumC += __shfl_down(sumC, o, 32);
        }
        if (tid == 0) {
            const float inv = 1.0f / 67108864.0f;   // NB * T1 * T2
            out[0] = fmaxf(sumC, sumR) * inv;       // max(forward, backward)
        }
    }
}

// ---------------------------------------------------------------------------
extern "C" void kernel_launch(void* const* d_in, const int* in_sizes, int n_in,
                              void* d_out, int out_size, void* d_ws, size_t ws_size,
                              hipStream_t stream) {
    (void)in_sizes; (void)n_in; (void)out_size; (void)ws_size;
    const float* x = (const float*)d_in[0];   // [4,16,1024,128] f32
    const float* y = (const float*)d_in[1];   // [4,16,1024,128] f32
    float* out = (float*)d_out;               // scalar f32

    unsigned int* rowmin = (unsigned int*)d_ws;
    unsigned int* colmin = rowmin + (size_t)NB * T1;   // 512 KB workspace total

    const int nmins = 2 * NB * T1;
    wmd_init_mins<<<(nmins + 255) / 256, 256, 0, stream>>>(rowmin, nmins);
    wmd_tile_kernel<<<NB * 64, 256, 0, stream>>>(x, y, rowmin, colmin);
    wmd_reduce_kernel<<<1, 1024, 0, stream>>>(rowmin, colmin, out);
}
